// GAT_Net_41979010351249
// MI455X (gfx1250) — matrix operand order repaired and checked by reading, not compiled, not run on hardware
//
#include <hip/hip_runtime.h>

typedef __attribute__((ext_vector_type(16))) _Float16 v16h;
typedef __attribute__((ext_vector_type(8)))  float    v8f;

#define N_NODES 100000
#define N_EDGES 1600000
#define E_FULL  (N_EDGES + N_NODES)   // self loops appended
#define HC      128                   // H * C
#define C_DIM   64
#define NEG_SLOPE 0.2f
#define ENC_NEG_INF 0x007FFFFFu       // order-preserving encoding of -inf

// ---- order-preserving float <-> uint for atomic max ----
__device__ __forceinline__ unsigned fenc(float f) {
  unsigned u = __float_as_uint(f);
  return (u & 0x80000000u) ? ~u : (u | 0x80000000u);
}
__device__ __forceinline__ float fdec(unsigned u) {
  unsigned b = (u & 0x80000000u) ? (u & 0x7FFFFFFFu) : ~u;
  return __uint_as_float(b);
}

// ---------------- scalar prep ----------------
__global__ void k_scalar_zero(float* scal) {
  if (threadIdx.x == 0) scal[0] = 0.f;
}

__global__ void k_ew_sum(const float* __restrict__ ew, float* scal) {
  __shared__ float sm[256];
  float s = 0.f;
  for (int i = blockIdx.x * blockDim.x + threadIdx.x; i < N_EDGES;
       i += gridDim.x * blockDim.x)
    s += ew[i];
  sm[threadIdx.x] = s;
  __syncthreads();
  for (int st = 128; st > 0; st >>= 1) {
    if ((int)threadIdx.x < st) sm[threadIdx.x] += sm[threadIdx.x + st];
    __syncthreads();
  }
  if (threadIdx.x == 0) atomicAdd(&scal[0], sm[0]);
}

// scal[0] = mean(ew); scal[1..2] = ec1[h]; scal[3..4] = ec2[h]
__global__ void k_coeffs(const float* __restrict__ we1, const float* __restrict__ ae1,
                         const float* __restrict__ we2, const float* __restrict__ ae2,
                         float* scal) {
  __shared__ float sm[128];
  int t = threadIdx.x;
  sm[t] = we1[t] * ae1[t];
  __syncthreads();
  if (t == 0) {
    float s0 = 0.f, s1 = 0.f;
    for (int i = 0; i < 64; ++i) { s0 += sm[i]; s1 += sm[64 + i]; }
    scal[1] = s0; scal[2] = s1;
  }
  __syncthreads();
  sm[t] = we2[t] * ae2[t];
  __syncthreads();
  if (t == 0) {
    float s0 = 0.f, s1 = 0.f;
    for (int i = 0; i < 64; ++i) { s0 += sm[i]; s1 += sm[64 + i]; }
    scal[3] = s0; scal[4] = s1;
    scal[0] = scal[0] / (float)N_EDGES;   // sum -> mean
  }
}

// ---------------- WMMA GEMM:  Out[m, j] = sum_k A[m,k] * W[j,k] ----------------
// A: [N_NODES, K] f32 row-major, W: [128, K] f32 row-major, Out: [N_NODES, 128] f32.
// One wave -> one 16x16 tile. blockIdx.x = M-tile (6250), wave-in-block = N-tile (8).
// K is a compile-time constant (64 or 128) so the K-loop fully unrolls into
// back-to-back v_wmma_f32_16x16x32_f16 with all loads clause-issued up front.
template <int K>
__global__ void k_gemm(const float* __restrict__ A,
                       const float* __restrict__ W, float* __restrict__ Out) {
  const int lane  = threadIdx.x & 31;
  const int tileN = threadIdx.x >> 5;     // 0..7
  const int tileM = blockIdx.x;           // 0..6249
  const int m  = lane & 15;
  const bool hi = lane >= 16;

  const float* arow = A + (size_t)(tileM * 16 + m) * K;   // A row (lane role: M)
  const float* wrow = W + (size_t)(tileN * 16 + m) * K;   // W row j (lane role: N col)

  v8f acc = {};
#pragma unroll
  for (int k0 = 0; k0 < K; k0 += 32) {
    // A fragment: lanes 0-15 hold K = k0+0..7 / k0+16..23 ; lanes 16-31 K = +8 / +24
    const int ka = k0 + (hi ? 8 : 0);
    float4 a0 = *(const float4*)(arow + ka);
    float4 a1 = *(const float4*)(arow + ka + 4);
    float4 a2 = *(const float4*)(arow + ka + 16);
    float4 a3 = *(const float4*)(arow + ka + 20);
    // B fragment: lanes 0-15 hold K = k0+0..7 / k0+16..23 ; lanes 16-31 K = +8 / +24
    const int kb = k0 + (hi ? 8 : 0);
    float4 b0 = *(const float4*)(wrow + kb);
    float4 b1 = *(const float4*)(wrow + kb + 4);
    float4 b2 = *(const float4*)(wrow + kb + 16);
    float4 b3 = *(const float4*)(wrow + kb + 20);

    v16h av, bv;
    av[0]=(_Float16)a0.x; av[1]=(_Float16)a0.y; av[2]=(_Float16)a0.z; av[3]=(_Float16)a0.w;
    av[4]=(_Float16)a1.x; av[5]=(_Float16)a1.y; av[6]=(_Float16)a1.z; av[7]=(_Float16)a1.w;
    av[8]=(_Float16)a2.x; av[9]=(_Float16)a2.y; av[10]=(_Float16)a2.z; av[11]=(_Float16)a2.w;
    av[12]=(_Float16)a3.x; av[13]=(_Float16)a3.y; av[14]=(_Float16)a3.z; av[15]=(_Float16)a3.w;
    bv[0]=(_Float16)b0.x; bv[1]=(_Float16)b0.y; bv[2]=(_Float16)b0.z; bv[3]=(_Float16)b0.w;
    bv[4]=(_Float16)b1.x; bv[5]=(_Float16)b1.y; bv[6]=(_Float16)b1.z; bv[7]=(_Float16)b1.w;
    bv[8]=(_Float16)b2.x; bv[9]=(_Float16)b2.y; bv[10]=(_Float16)b2.z; bv[11]=(_Float16)b2.w;
    bv[12]=(_Float16)b3.x; bv[13]=(_Float16)b3.y; bv[14]=(_Float16)b3.z; bv[15]=(_Float16)b3.w;

    acc = __builtin_amdgcn_wmma_f32_16x16x32_f16(false, av, false, bv,
                                                 (short)0, acc, false, false);
  }
  // C/D layout: VGPR r -> lanes 0-15: M=r, N=lane ; lanes 16-31: M=8+r, N=lane-16
  const int col = tileN * 16 + m;
#pragma unroll
  for (int r = 0; r < 8; ++r) {
    int row = tileM * 16 + (hi ? r + 8 : r);
    Out[(size_t)row * HC + col] = acc[r];
  }
}

// ---------------- per-node attention logits ----------------
__global__ void k_node_alpha(const float* __restrict__ Hm,
                             const float* __restrict__ a_src,
                             const float* __restrict__ a_dst,
                             float* __restrict__ al_s, float* __restrict__ al_d) {
  int idx = blockIdx.x * blockDim.x + threadIdx.x;
  if (idx >= N_NODES * 2) return;
  int n = idx >> 1, h = idx & 1;
  const float* hr = Hm + (size_t)n * HC + h * C_DIM;
  const float* as = a_src + h * C_DIM;
  const float* ad = a_dst + h * C_DIM;
  float ss = 0.f, sd = 0.f;
  for (int c = 0; c < C_DIM; c += 4) {
    float4 hv = *(const float4*)(hr + c);
    float4 av = *(const float4*)(as + c);
    float4 dv = *(const float4*)(ad + c);
    ss += hv.x*av.x + hv.y*av.y + hv.z*av.z + hv.w*av.w;
    sd += hv.x*dv.x + hv.y*dv.y + hv.z*dv.z + hv.w*dv.w;
  }
  al_s[idx] = ss;
  al_d[idx] = sd;
}

// ---------------- init accumulators for a layer ----------------
__global__ void k_init(float* __restrict__ acc, unsigned* __restrict__ m_enc,
                       float* __restrict__ s) {
  int idx = blockIdx.x * blockDim.x + threadIdx.x;
  if (idx < N_NODES * HC) acc[idx] = 0.f;
  if (idx < N_NODES * 2) { m_enc[idx] = ENC_NEG_INF; s[idx] = 0.f; }
}

__device__ __forceinline__ void edge_get(int e, const int* __restrict__ ei,
                                         const float* __restrict__ ew, float mean_ew,
                                         int& si, int& di, float& we) {
  if (e < N_EDGES) { si = ei[e]; di = ei[N_EDGES + e]; we = ew[e]; }
  else             { si = di = e - N_EDGES; we = mean_ew; }
}

// ---------------- softmax pass 1: segment max ----------------
__global__ void k_edge_max(const int* __restrict__ ei, const float* __restrict__ ew,
                           const float* __restrict__ scal, const float* __restrict__ ec,
                           const float* __restrict__ al_s, const float* __restrict__ al_d,
                           unsigned* __restrict__ m_enc) {
  int t = blockIdx.x * blockDim.x + threadIdx.x;
  if (t >= E_FULL * 2) return;
  int e = t >> 1, h = t & 1;
  int si, di; float we;
  edge_get(e, ei, ew, scal[0], si, di, we);
  float a = al_s[si * 2 + h] + al_d[di * 2 + h] + ec[h] * we;
  a = (a > 0.f) ? a : NEG_SLOPE * a;               // leaky relu
  atomicMax(&m_enc[di * 2 + h], fenc(a));
}

// ---------------- softmax pass 2: exp + segment sum ----------------
__global__ void k_edge_sum(const int* __restrict__ ei, const float* __restrict__ ew,
                           const float* __restrict__ scal, const float* __restrict__ ec,
                           const float* __restrict__ al_s, const float* __restrict__ al_d,
                           const unsigned* __restrict__ m_enc,
                           float* __restrict__ ex_buf, float* __restrict__ ssum) {
  int t = blockIdx.x * blockDim.x + threadIdx.x;
  if (t >= E_FULL * 2) return;
  int e = t >> 1, h = t & 1;
  int si, di; float we;
  edge_get(e, ei, ew, scal[0], si, di, we);
  float a = al_s[si * 2 + h] + al_d[di * 2 + h] + ec[h] * we;
  a = (a > 0.f) ? a : NEG_SLOPE * a;
  float m = fdec(m_enc[di * 2 + h]);
  float ex = expf(a - m);
  ex_buf[t] = ex;
  atomicAdd(&ssum[di * 2 + h], ex);
}

// ---------------- pass 3: weighted scatter of h[src] into acc[dst] ----------------
// 32 threads per edge: 2 heads x 16 float4 channel-chunks.
__global__ void k_edge_agg(const int* __restrict__ ei, const float* __restrict__ ex_buf,
                           const float* __restrict__ ssum, const float* __restrict__ Hm,
                           float* __restrict__ acc) {
  long long t = (long long)blockIdx.x * blockDim.x + threadIdx.x;
  if (t >= (long long)E_FULL * 32) return;
  int e   = (int)(t >> 5);
  int rem = (int)(t & 31);
  int h   = rem >> 4;
  int c4  = (rem & 15) * 4;
  int si, di;
  if (e < N_EDGES) { si = ei[e]; di = ei[N_EDGES + e]; }
  else             { si = di = e - N_EDGES; }
  float w = ex_buf[e * 2 + h] / (ssum[di * 2 + h] + 1e-16f);
  float4 hv = *(const float4*)(Hm + (size_t)si * HC + h * C_DIM + c4);
  float* ap = acc + (size_t)di * HC + h * C_DIM + c4;
  atomicAdd(ap + 0, hv.x * w);
  atomicAdd(ap + 1, hv.y * w);
  atomicAdd(ap + 2, hv.z * w);
  atomicAdd(ap + 3, hv.w * w);
}

// ---------------- layer-1 activation: x1 = relu(acc + bias1), in place ----------------
__global__ void k_final1(float* __restrict__ x1, const float* __restrict__ bias) {
  int idx = blockIdx.x * blockDim.x + threadIdx.x;
  if (idx >= N_NODES * HC) return;
  float v = x1[idx] + bias[idx & 127];
  x1[idx] = v > 0.f ? v : 0.f;
}

// ---------------- layer-2 head mean + bias -> output ----------------
__global__ void k_final2(const float* __restrict__ acc2, const float* __restrict__ bias,
                         float* __restrict__ out) {
  int idx = blockIdx.x * blockDim.x + threadIdx.x;
  if (idx >= N_NODES * C_DIM) return;
  int n = idx >> 6, c = idx & 63;
  out[idx] = 0.5f * (acc2[(size_t)n * HC + c] + acc2[(size_t)n * HC + 64 + c]) + bias[c];
}

extern "C" void kernel_launch(void* const* d_in, const int* in_sizes, int n_in,
                              void* d_out, int out_size, void* d_ws, size_t ws_size,
                              hipStream_t stream) {
  (void)in_sizes; (void)n_in; (void)out_size; (void)ws_size;
  const float* x   = (const float*)d_in[0];
  const int*   ei  = (const int*)  d_in[1];
  const float* ew  = (const float*)d_in[2];
  const float* w1  = (const float*)d_in[3];
  const float* as1 = (const float*)d_in[4];
  const float* ad1 = (const float*)d_in[5];
  const float* we1 = (const float*)d_in[6];
  const float* ae1 = (const float*)d_in[7];
  const float* b1  = (const float*)d_in[8];
  const float* w2  = (const float*)d_in[9];
  const float* as2 = (const float*)d_in[10];
  const float* ad2 = (const float*)d_in[11];
  const float* we2 = (const float*)d_in[12];
  const float* ae2 = (const float*)d_in[13];
  const float* b2  = (const float*)d_in[14];
  float* out = (float*)d_out;

  // workspace carve-up (256B aligned slices)
  char* base = (char*)d_ws;
  size_t off = 0;
  auto carve = [&](size_t bytes) -> void* {
    void* p = base + off;
    off += (bytes + 255) & ~(size_t)255;
    return p;
  };
  float*    scal  = (float*)   carve(64 * sizeof(float));
  float*    hbuf  = (float*)   carve((size_t)N_NODES * HC * sizeof(float)); // h (layer in use)
  float*    x1buf = (float*)   carve((size_t)N_NODES * HC * sizeof(float)); // layer1 acc -> x1
  float*    acc2  = (float*)   carve((size_t)N_NODES * HC * sizeof(float)); // layer2 acc
  float*    al_s  = (float*)   carve((size_t)N_NODES * 2 * sizeof(float));
  float*    al_d  = (float*)   carve((size_t)N_NODES * 2 * sizeof(float));
  unsigned* m_enc = (unsigned*)carve((size_t)N_NODES * 2 * sizeof(unsigned));
  float*    ssum  = (float*)   carve((size_t)N_NODES * 2 * sizeof(float));
  float*    exbuf = (float*)   carve((size_t)E_FULL * 2 * sizeof(float));

  const int B = 256;
  const int gNode2 = (N_NODES * 2 + B - 1) / B;
  const int gNodeF = (N_NODES * HC + B - 1) / B;
  const int gEdge2 = (E_FULL * 2 + B - 1) / B;
  const int gEdge32 = (E_FULL * 32 + B - 1) / B;

  // scalar prep: mean(edge_weight), edge-attention coefficients
  k_scalar_zero<<<1, 64, 0, stream>>>(scal);
  k_ew_sum<<<1024, B, 0, stream>>>(ew, scal);
  k_coeffs<<<1, 128, 0, stream>>>(we1, ae1, we2, ae2, scal);

  // ---------- layer 1 ----------
  k_gemm<64><<<N_NODES / 16, B, 0, stream>>>(x, w1, hbuf);
  k_node_alpha<<<gNode2, B, 0, stream>>>(hbuf, as1, ad1, al_s, al_d);
  k_init<<<gNodeF, B, 0, stream>>>(x1buf, m_enc, ssum);
  k_edge_max<<<gEdge2, B, 0, stream>>>(ei, ew, scal, scal + 1, al_s, al_d, m_enc);
  k_edge_sum<<<gEdge2, B, 0, stream>>>(ei, ew, scal, scal + 1, al_s, al_d, m_enc, exbuf, ssum);
  k_edge_agg<<<gEdge32, B, 0, stream>>>(ei, exbuf, ssum, hbuf, x1buf);
  k_final1<<<gNodeF, B, 0, stream>>>(x1buf, b1);

  // ---------- layer 2 ----------
  k_gemm<128><<<N_NODES / 16, B, 0, stream>>>(x1buf, w2, hbuf);
  k_node_alpha<<<gNode2, B, 0, stream>>>(hbuf, as2, ad2, al_s, al_d);
  k_init<<<gNodeF, B, 0, stream>>>(acc2, m_enc, ssum);
  k_edge_max<<<gEdge2, B, 0, stream>>>(ei, ew, scal, scal + 3, al_s, al_d, m_enc);
  k_edge_sum<<<gEdge2, B, 0, stream>>>(ei, ew, scal, scal + 3, al_s, al_d, m_enc, exbuf, ssum);
  k_edge_agg<<<gEdge32, B, 0, stream>>>(ei, exbuf, ssum, hbuf, acc2);
  k_final2<<<(N_NODES * C_DIM + B - 1) / B, B, 0, stream>>>(acc2, b2, out);
}
